// HeteroGraphSAGE_LINK_24524263260184
// MI455X (gfx1250) — matrix operand order repaired and checked by reading, not compiled, not run on hardware
//
#include <hip/hip_runtime.h>

#define N_A  50000
#define N_B  50000
#define NT   100000
#define NE   800000
#define EHN  1600000
#define CCH  128
#define PEIN 8
#define HID  64
#define PD   64

typedef __bf16 v16bf __attribute__((ext_vector_type(16)));
typedef float  v8f   __attribute__((ext_vector_type(8)));

union Frag { v16bf v; unsigned short u[16]; uint4 q[2]; };

__device__ __forceinline__ unsigned short f2bf(float f) {
  unsigned int u = __builtin_bit_cast(unsigned int, f);
  u += 0x7fffu + ((u >> 16) & 1u);            // round-to-nearest-even
  return (unsigned short)(u >> 16);
}
__device__ __forceinline__ float bf2f(unsigned short h) {
  unsigned int u = ((unsigned int)h) << 16;
  return __builtin_bit_cast(float, u);
}

// ---------------- utility kernels ----------------
__global__ void k_zero_f32(float* __restrict__ p, int n) {
  int i = blockIdx.x * blockDim.x + threadIdx.x;
  if (i < n) p[i] = 0.0f;
}
__global__ void k_copy_f32(const float* __restrict__ s, float* __restrict__ d, int n) {
  int i = blockIdx.x * blockDim.x + threadIdx.x;
  if (i < n) d[i] = s[i];
}
__global__ void k_f32_to_bf16(const float* __restrict__ s, unsigned short* __restrict__ d, int n) {
  int i = blockIdx.x * blockDim.x + threadIdx.x;
  if (i < n) d[i] = f2bf(s[i]);
}

// Pack weights [K x 128] f32 (rowsA rows from srcA, then K-rowsA rows from srcB)
// into WMMA B-fragment order, bf16:
//   dst[(((c*2 + h)*128 + n)*16) + t] = W[c*32 + h*16 + t][n]
// so each lane's 16 halfs for a K=32 chunk are contiguous (2x ds_load_b128).
__global__ void k_pack_w(const float* __restrict__ srcA, const float* __restrict__ srcB,
                         int rowsA, unsigned short* __restrict__ dst, int K) {
  int o = blockIdx.x * blockDim.x + threadIdx.x;
  if (o >= K * 128) return;
  int t  = o & 15;
  int n  = (o >> 4) & 127;
  int ch = o >> 11;                 // c*2 + h
  int row = (ch >> 1) * 32 + (ch & 1) * 16 + t;
  float v = (row < rowsA) ? srcA[(long)row * 128 + n]
                          : srcB[(long)(row - rowsA) * 128 + n];
  dst[o] = f2bf(v);
}

// ---------------- homogeneous PE aggregation: agg = PE + seg_sum(PE[hsrc] -> hdst) ----
__global__ void k_pe_agg(const int* __restrict__ hsrc, const int* __restrict__ hdst,
                         const float* __restrict__ PE, float* __restrict__ agg) {
  int i = blockIdx.x * blockDim.x + threadIdx.x;
  if (i >= EHN * PEIN) return;
  int e  = i >> 3;
  int ch = i & 7;
  int s = hsrc[e], d = hdst[e];
  atomicAdd(&agg[(long)d * PEIN + ch], PE[(long)s * PEIN + ch]);
}

// ---------------- phi MLP: pe = relu(agg@W1+b1)@W2+b2  (8->64->64), bf16 out -----
__global__ __launch_bounds__(256) void k_phi(const float* __restrict__ agg,
                                             const float* __restrict__ W1, const float* __restrict__ b1,
                                             const float* __restrict__ W2, const float* __restrict__ b2,
                                             unsigned short* __restrict__ pe_out) {
  __shared__ float sW1[PEIN * HID];
  __shared__ float sW2[HID * PD];
  __shared__ float sb1[HID];
  __shared__ float sb2[PD];
  for (int i = threadIdx.x; i < PEIN * HID; i += 256) sW1[i] = W1[i];
  for (int i = threadIdx.x; i < HID * PD;  i += 256) sW2[i] = W2[i];
  if (threadIdx.x < HID) sb1[threadIdx.x] = b1[threadIdx.x];
  if (threadIdx.x < PD)  sb2[threadIdx.x] = b2[threadIdx.x];
  __syncthreads();
  int node = blockIdx.x * 256 + threadIdx.x;
  if (node >= NT) return;
  float a[PEIN];
#pragma unroll
  for (int k = 0; k < PEIN; ++k) a[k] = agg[(long)node * PEIN + k];
  float h[HID];
  for (int j = 0; j < HID; ++j) {
    float acc = sb1[j];
#pragma unroll
    for (int k = 0; k < PEIN; ++k) acc += a[k] * sW1[k * HID + j];
    h[j] = fmaxf(acc, 0.0f);
  }
  for (int j = 0; j < PD; ++j) {
    float acc = sb2[j];
    for (int k = 0; k < HID; ++k) acc += h[k] * sW2[k * PD + j];
    pe_out[(long)node * PD + j] = f2bf(acc);
  }
}

// ---------------- edge scatter: sum[dst] += bf16 X[src], cnt[dst] += 1 -----------
__global__ void k_scatter(const int* __restrict__ src, const int* __restrict__ dst,
                          const unsigned short* __restrict__ X,
                          float* __restrict__ sum, float* __restrict__ cnt) {
  int gid = blockIdx.x * blockDim.x + threadIdx.x;
  if (gid >= NE * 32) return;
  int e = gid >> 5, sub = gid & 31;
  int s = src[e], d = dst[e];
  int ch = sub * 4;
  const unsigned short* xp = X + (long)s * CCH + ch;
  uint2 pk = *(const uint2*)xp;
  float v0 = bf2f((unsigned short)(pk.x & 0xffffu));
  float v1 = bf2f((unsigned short)(pk.x >> 16));
  float v2 = bf2f((unsigned short)(pk.y & 0xffffu));
  float v3 = bf2f((unsigned short)(pk.y >> 16));
  float* sp = sum + (long)d * CCH + ch;
  atomicAdd(sp + 0, v0);
  atomicAdd(sp + 1, v1);
  atomicAdd(sp + 2, v2);
  atomicAdd(sp + 3, v3);
  if (sub == 0) atomicAdd(&cnt[d], 1.0f);
}

__global__ void k_mean(const float* __restrict__ sum, const float* __restrict__ cnt,
                       unsigned short* __restrict__ mbar, int nnodes) {
  int i = blockIdx.x * blockDim.x + threadIdx.x;
  if (i >= nnodes * CCH) return;
  int node = i >> 7;
  float c = fmaxf(cnt[node], 1.0f);
  mbar[i] = f2bf(sum[i] / c);
}

// ---------------- fused WMMA GEMM: out = concat(X1,X2) @ W + bias [-> LN -> relu] ----
// Block = 256 threads = 8 waves; block computes a 64-row x 128-col tile.
// Wave w owns the 16-col strip at w*16 and computes 4 row-tiles, reusing one
// B fragment per K-step for 4 WMMAs (v_wmma_f32_16x16x32_bf16).
// W must be in k_pack_w fragment order.
__global__ __launch_bounds__(256) void k_gemm(
    const unsigned short* __restrict__ X1, int k1,
    const unsigned short* __restrict__ X2, int k2,
    const unsigned short* __restrict__ W,     // packed [(K/32) chunks][2][128][16] bf16
    const float* __restrict__ bias,           // [128]
    const float* __restrict__ gamma, const float* __restrict__ beta,
    int do_ln,
    unsigned short* __restrict__ out_bf,      // [M x 128] bf16
    float* __restrict__ out_f32,              // optional [M x 128] f32 (may be null)
    int M) {
  const int K = k1 + k2;                      // 192 or 256, multiple of 64
  // A tile (64 x K bf16, 32KB max) aliased with LN tile (64 x 128 f32, 32KB):
  // A is dead before Ct is written; a barrier separates the two uses.
  __shared__ __align__(16) char ACbuf[64 * 256 * 2];
  unsigned short* Alds = (unsigned short*)ACbuf;
  float*          Ct   = (float*)ACbuf;
  __shared__ unsigned short Wlds[64 * 128];   // packed W chunk (64 K-rows)
  __shared__ float s_mu[64], s_rs[64];

  const int tid = threadIdx.x;
  const int r0  = blockIdx.x * 64;

  // stage A tile (concat of X1|X2 per row), uint4-vectorized
  const int KD = K >> 3;                      // uint4 per row
  const uint4* X1q = (const uint4*)X1;
  const uint4* X2q = (const uint4*)X2;
  for (int idx = tid; idx < 64 * KD; idx += 256) {
    int r = idx / KD, cq = idx - r * KD;
    int c = cq << 3;
    int row = r0 + r;
    uint4 v = make_uint4(0u, 0u, 0u, 0u);
    if (row < M) v = (c < k1) ? X1q[(long)row * (k1 >> 3) + cq]
                              : X2q[(long)row * (k2 >> 3) + ((c - k1) >> 3)];
    *(uint4*)(Alds + r * K + c) = v;
  }

  const int lane = tid & 31;
  const int wave = tid >> 5;
  const int mrow = lane & 15;     // A row within 16-row tile
  const int hsel = lane >> 4;     // lane half select
  const int ncol = wave * 16 + (lane & 15);

  v8f acc[4];
#pragma unroll
  for (int t = 0; t < 4; ++t) acc[t] = {};

  for (int kb0 = 0; kb0 < K; kb0 += 64) {
    __syncthreads();
    {   // copy packed 64-row W chunk into LDS (contiguous, uint4-vectorized)
      const uint4* Wg = (const uint4*)(W + (long)kb0 * 128);
      uint4* Wl = (uint4*)Wlds;
      for (int i = tid; i < 64 * 128 / 8; i += 256) Wl[i] = Wg[i];
    }
    __syncthreads();
#pragma unroll
    for (int kk = 0; kk < 64; kk += 32) {
      // B 32x16 bf16 fragment: contiguous 16-half run in packed layout (shared by 4 tiles)
      Frag bfr;
      int fb = (((kk >> 5) * 2 + hsel) * 128 + ncol) * 16;
      bfr.q[0] = *(const uint4*)(Wlds + fb);
      bfr.q[1] = *(const uint4*)(Wlds + fb + 8);
      int abase = kb0 + kk + hsel * 16;
#pragma unroll
      for (int t = 0; t < 4; ++t) {
        // A 16x32 bf16 fragment: one contiguous 16-half run per lane
        Frag af;
        int a0 = (t * 16 + mrow) * K + abase;
        af.q[0] = *(const uint4*)(Alds + a0);
        af.q[1] = *(const uint4*)(Alds + a0 + 8);
        acc[t] = __builtin_amdgcn_wmma_f32_16x16x32_bf16(false, af.v, false, bfr.v,
                                                         (short)0, acc[t], false, false);
      }
    }
  }

  const float bv = bias[ncol];
  if (!do_ln) {
#pragma unroll
    for (int t = 0; t < 4; ++t) {
#pragma unroll
      for (int j = 0; j < 8; ++j) {
        int row = r0 + t * 16 + hsel * 8 + j;   // C/D: lanes 0-15 -> M=j, lanes 16-31 -> M=8+j
        if (row < M) {
          float val = acc[t][j] + bv;
          out_bf[(long)row * 128 + ncol] = f2bf(val);
          if (out_f32) out_f32[(long)row * 128 + ncol] = val;
        }
      }
    }
  } else {
    __syncthreads();                            // A tile dead; reuse buffer as Ct
#pragma unroll
    for (int t = 0; t < 4; ++t)
#pragma unroll
      for (int j = 0; j < 8; ++j)
        Ct[(t * 16 + hsel * 8 + j) * 128 + ncol] = acc[t][j] + bv;
    __syncthreads();
    if (tid < 64) {
      float s = 0.0f, s2 = 0.0f;
      for (int c = 0; c < 128; ++c) { float x = Ct[tid * 128 + c]; s += x; s2 += x * x; }
      float mu = s * (1.0f / 128.0f);
      float var = s2 * (1.0f / 128.0f) - mu * mu;
      s_mu[tid] = mu;
      s_rs[tid] = rsqrtf(var + 1e-5f);
    }
    __syncthreads();
    for (int idx = tid; idx < 64 * 128; idx += 256) {
      int r = idx >> 7, c = idx & 127;
      int row = r0 + r;
      if (row < M) {
        float x = (Ct[idx] - s_mu[r]) * s_rs[r] * gamma[c] + beta[c];
        x = fmaxf(x, 0.0f);
        out_bf[(long)row * 128 + c] = f2bf(x);
        if (out_f32) out_f32[(long)row * 128 + c] = x;
      }
    }
  }
}

// ---------------- host launcher ----------------
#define GRID(n, b) (((n) + (b) - 1) / (b))

extern "C" void kernel_launch(void* const* d_in, const int* in_sizes, int n_in,
                              void* d_out, int out_size, void* d_ws, size_t ws_size,
                              hipStream_t stream) {
  (void)in_sizes; (void)n_in; (void)out_size; (void)ws_size;
  const float* xA  = (const float*)d_in[0];
  const float* xB  = (const float*)d_in[1];
  const float* PE  = (const float*)d_in[2];
  const int*   eAB = (const int*)d_in[3];
  const int*   eBA = (const int*)d_in[4];
  const int*   he  = (const int*)d_in[5];
  const float* sWl = (const float*)d_in[6];
  const float* sbl = (const float*)d_in[7];
  const float* sWr = (const float*)d_in[8];
  const float* lng = (const float*)d_in[9];
  const float* lnb = (const float*)d_in[10];
  const float* pW1 = (const float*)d_in[11];
  const float* pb1 = (const float*)d_in[12];
  const float* pW2 = (const float*)d_in[13];
  const float* pb2 = (const float*)d_in[14];
  const float* peW = (const float*)d_in[15];
  const float* peb = (const float*)d_in[16];

  char* ws = (char*)d_ws;
  size_t off = 0;
  auto alloc = [&](size_t bytes) -> char* {
    char* p = ws + off;
    off += (bytes + 255) & ~(size_t)255;
    return p;
  };

  float*          agg   = (float*)alloc((size_t)NT * PEIN * 4);
  unsigned short* pe_bf = (unsigned short*)alloc((size_t)NT * PD * 2);
  unsigned short* xAbf  = (unsigned short*)alloc((size_t)N_A * CCH * 2);
  unsigned short* xBbf  = (unsigned short*)alloc((size_t)N_B * CCH * 2);
  unsigned short* xcA   = (unsigned short*)alloc((size_t)N_A * CCH * 2);
  unsigned short* xcB   = (unsigned short*)alloc((size_t)N_B * CCH * 2);
  float*          sumA  = (float*)alloc((size_t)N_A * CCH * 4);
  float*          sumB  = (float*)alloc((size_t)N_B * CCH * 4);
  float*          cntA  = (float*)alloc((size_t)N_A * 4);
  float*          cntB  = (float*)alloc((size_t)N_B * 4);
  unsigned short* mAbf  = (unsigned short*)alloc((size_t)N_A * CCH * 2);
  unsigned short* mBbf  = (unsigned short*)alloc((size_t)N_B * CCH * 2);
  unsigned short* peWpk = (unsigned short*)alloc((size_t)2 * 192 * 128 * 2);
  unsigned short* wcat  = (unsigned short*)alloc((size_t)4 * 256 * 128 * 2); // [l][t] packed

  // -------- one-time prep (per launch; deterministic) --------
  k_copy_f32<<<GRID(NT * PEIN, 256), 256, 0, stream>>>(PE, agg, NT * PEIN);
  k_pe_agg<<<GRID(EHN * PEIN, 256), 256, 0, stream>>>(he, he + EHN, PE, agg);
  k_f32_to_bf16<<<GRID(N_A * CCH, 256), 256, 0, stream>>>(xA, xAbf, N_A * CCH);
  k_f32_to_bf16<<<GRID(N_B * CCH, 256), 256, 0, stream>>>(xB, xBbf, N_B * CCH);
  for (int l = 0; l < 2; ++l) {
    // pe_W[l]: 192x128, all rows from one source
    k_pack_w<<<GRID(192 * 128, 256), 256, 0, stream>>>(peW + (size_t)l * 192 * 128,
                                                       peW + (size_t)l * 192 * 128, 192,
                                                       peWpk + (size_t)l * 192 * 128, 192);
    for (int t = 0; t < 2; ++t) {
      // [Wl ; Wr] stacked as 256x128
      k_pack_w<<<GRID(256 * 128, 256), 256, 0, stream>>>(sWl + (size_t)(l * 2 + t) * 16384,
                                                         sWr + (size_t)(l * 2 + t) * 16384, 128,
                                                         wcat + (size_t)(l * 2 + t) * 256 * 128, 256);
    }
  }

  float* outA = (float*)d_out;
  float* outB = (float*)d_out + (size_t)N_A * CCH;

  for (int l = 0; l < 2; ++l) {
    const int last = (l == 1);
    // phi MLP -> pe (bf16, [NT x 64])
    k_phi<<<GRID(NT, 256), 256, 0, stream>>>(agg, pW1 + (size_t)l * PEIN * HID, pb1 + (size_t)l * HID,
                                             pW2 + (size_t)l * HID * PD, pb2 + (size_t)l * PD, pe_bf);
    // concat linear: x' = [x | pe] @ pe_W + pe_b   (K = 192)
    k_gemm<<<GRID(N_A, 64), 256, 0, stream>>>(xAbf, 128, pe_bf, 64,
                                              peWpk + (size_t)l * 192 * 128, peb + (size_t)l * 128,
                                              nullptr, nullptr, 0, xcA, nullptr, N_A);
    k_gemm<<<GRID(N_B, 64), 256, 0, stream>>>(xBbf, 128, pe_bf + (size_t)N_A * PD, 64,
                                              peWpk + (size_t)l * 192 * 128, peb + (size_t)l * 128,
                                              nullptr, nullptr, 0, xcB, nullptr, N_B);
    // seg_mean A->B
    k_zero_f32<<<GRID(N_B * CCH, 256), 256, 0, stream>>>(sumB, N_B * CCH);
    k_zero_f32<<<GRID(N_B, 256), 256, 0, stream>>>(cntB, N_B);
    k_scatter<<<GRID(NE * 32, 256), 256, 0, stream>>>(eAB, eAB + NE, xcA, sumB, cntB);
    k_mean<<<GRID(N_B * CCH, 256), 256, 0, stream>>>(sumB, cntB, mBbf, N_B);
    // seg_mean B->A
    k_zero_f32<<<GRID(N_A * CCH, 256), 256, 0, stream>>>(sumA, N_A * CCH);
    k_zero_f32<<<GRID(N_A, 256), 256, 0, stream>>>(cntA, N_A);
    k_scatter<<<GRID(NE * 32, 256), 256, 0, stream>>>(eBA, eBA + NE, xcB, sumA, cntA);
    k_mean<<<GRID(N_A * CCH, 256), 256, 0, stream>>>(sumA, cntA, mAbf, N_A);
    // SAGE: new = [mean | x] @ [Wl;Wr] + bl -> LN -> relu   (K = 256)
    k_gemm<<<GRID(N_B, 64), 256, 0, stream>>>(mBbf, 128, xcB, 128,
                                              wcat + (size_t)(l * 2 + 0) * 256 * 128, sbl + (size_t)(l * 2 + 0) * 128,
                                              lng + (size_t)(l * 2 + 1) * 128, lnb + (size_t)(l * 2 + 1) * 128, 1,
                                              xBbf, last ? outB : nullptr, N_B);
    k_gemm<<<GRID(N_A, 64), 256, 0, stream>>>(mAbf, 128, xcA, 128,
                                              wcat + (size_t)(l * 2 + 1) * 256 * 128, sbl + (size_t)(l * 2 + 1) * 128,
                                              lng + (size_t)(l * 2 + 0) * 128, lnb + (size_t)(l * 2 + 0) * 128, 1,
                                              xAbf, last ? outA : nullptr, N_A);
  }
}
